// FFN_46961172414970
// MI455X (gfx1250) — matrix operand order repaired and checked by reading, not compiled, not run on hardware
//
#include <hip/hip_runtime.h>
#include <hip/hip_bf16.h>

typedef __attribute__((ext_vector_type(16))) _Float16 v16h;
typedef __attribute__((ext_vector_type(8)))  _Float16 v8h;
typedef __attribute__((ext_vector_type(8)))  float    v8f;

#define WPB 4
#define TPB (WPB * 32)

// fragment table layout in d_ws (each frag = 32 lanes * 16 halfs = 1KB)
#define FR_WE0 0
#define FR_WE1 1
#define FR_WE2 2
#define FR_WE3 4
#define FR_WA0 8
#define FR_WO0 12
#define FR_WO1 28
#define NFRAG  92
#define AGG_OFF (128 * 1024)   // frags use 92KB; agg starts at 128KB

__device__ __forceinline__ v8f wmma_f16(v16h a, v16h b, v8f c) {
  return __builtin_amdgcn_wmma_f32_16x16x32_f16(false, a, false, b, (short)0, c,
                                                false, false);
}

__device__ __forceinline__ v16h cat8(v8h lo, v8h hi) {
  return __builtin_shufflevector(lo, hi, 0, 1, 2, 3, 4, 5, 6, 7, 8, 9, 10, 11,
                                 12, 13, 14, 15);
}

// B fragment: lane = column n (mod 16), half-group g holds K=8g..8g+7 then K=16+8g..16+8g+7
__device__ __forceinline__ v16h load_b_frag(const _Float16* __restrict__ wsB,
                                            int frag, int lane) {
  const _Float16* p = wsB + (size_t)frag * 512 + lane * 16;
  v8h lo = *(const v8h*)p;
  v8h hi = *(const v8h*)(p + 8);
  return cat8(lo, hi);
}

// A fragment from an LDS f16 tile: lane = row m, halves j0..7 <-> K=8g+j,
// j8..15 <-> K=16+8g+(j-8)
__device__ __forceinline__ v16h load_a_frag(const _Float16* H, int lane,
                                            int stride, int colbase) {
  int m = lane & 15, g = lane >> 4;
  const _Float16* p0 = H + m * stride + colbase + 8 * g;
  v8h lo = *(const v8h*)p0;
  v8h hi = *(const v8h*)(p0 + 16);
  return cat8(lo, hi);
}

// D(f32) tile -> f16 LDS tile: lane holds column n, rows m = 8g + r
__device__ __forceinline__ void store_d_frag(_Float16* H, int lane, int stride,
                                             int colbase, v8f d) {
  int n = lane & 15, g = lane >> 4;
#pragma unroll
  for (int r = 0; r < 8; ++r)
    H[(8 * g + r) * stride + colbase + n] = (_Float16)d[r];
}

// bias (pre-loaded per-lane scalar) + relu
__device__ __forceinline__ v8f bias_relu_v(v8f d, float b) {
  v8f o;
#pragma unroll
  for (int r = 0; r < 8; ++r) o[r] = fmaxf(d[r] + b, 0.f);
  return o;
}

// ---------------------------------------------------------------------------
// Pack all weights to f16 WMMA B-fragments (zero-padded K->32, N->16).
__global__ void pack_weights(const float* __restrict__ we0,
                             const float* __restrict__ we1,
                             const float* __restrict__ we2,
                             const float* __restrict__ we3,
                             const float* __restrict__ wa0,
                             const float* __restrict__ wo0,
                             const float* __restrict__ wo1,
                             _Float16* __restrict__ wsB) {
  int f = blockIdx.x;
  int lane = threadIdx.x;
  const float* W;
  int din, dout, kb, nb;
  if (f == 0)      { W = we0; din = 11;  dout = 8;   kb = 0; nb = 0; }
  else if (f == 1) { W = we1; din = 8;   dout = 16;  kb = 0; nb = 0; }
  else if (f < 4)  { W = we2; din = 16;  dout = 32;  kb = 0; nb = f - 2; }
  else if (f < 8)  { W = we3; din = 32;  dout = 64;  kb = 0; nb = f - 4; }
  else if (f < 12) { W = wa0; din = 11;  dout = 64;  kb = 0; nb = f - 8; }
  else if (f < 28) { int i = f - 12; W = wo0; din = 64;  dout = 128; kb = i >> 3; nb = i & 7;  }
  else             { int i = f - 28; W = wo1; din = 128; dout = 256; kb = i >> 4; nb = i & 15; }
  int g = lane >> 4, n = lane & 15;
  _Float16* dst = wsB + (size_t)f * 512 + lane * 16;
#pragma unroll
  for (int j = 0; j < 16; ++j) {
    int k = kb * 32 + (j < 8 ? 8 * g + j : 16 + 8 * g + (j - 8));
    int ng = nb * 16 + n;
    float v = (k < din && ng < dout) ? W[(size_t)k * dout + ng] : 0.f;
    dst[j] = (_Float16)v;
  }
}

__global__ void zero_f32(float* __restrict__ p, size_t n) {
  size_t i = (size_t)blockIdx.x * blockDim.x + threadIdx.x;
  size_t stride = (size_t)gridDim.x * blockDim.x;
  for (; i < n; i += stride) p[i] = 0.f;
}

// ---------------------------------------------------------------------------
// Per-point edge MLP + attention + scatter-sum into agg[C,64].
__global__ void __launch_bounds__(TPB) edge_attn_scatter(
    const float* __restrict__ pf, const int* __restrict__ labels,
    const float* __restrict__ cc, const float* __restrict__ pts,
    const float* __restrict__ be0, const float* __restrict__ be1,
    const float* __restrict__ be2, const float* __restrict__ be3,
    const float* __restrict__ ba0, const float* __restrict__ wa1,
    const float* __restrict__ ba1, const _Float16* __restrict__ wsB,
    float* __restrict__ agg, int npts) {
  __shared__ __align__(16) _Float16 Hs[WPB][16 * 32];
  __shared__ int Ls[WPB][16];
  const int lane = threadIdx.x & 31;
  const int wave = threadIdx.x >> 5;
  const int g = lane >> 4, nn = lane & 15;
  _Float16* H = Hs[wave];
  int* LB = Ls[wave];
  const int ntiles = (npts + 15) >> 4;
  const int wavesTotal = gridDim.x * WPB;

  // ---- hoist all loop-invariant state into registers -----------------------
  const float ba1v = ba1[0];
  float wv[4];
#pragma unroll
  for (int t = 0; t < 4; ++t) wv[t] = wa1[t * 16 + nn];
  const float b0 = (nn < 8) ? be0[nn] : 0.f;
  const float b1 = be1[nn];
  float b2[2], b3[4], ba[4];
#pragma unroll
  for (int t = 0; t < 2; ++t) b2[t] = be2[t * 16 + nn];
#pragma unroll
  for (int t = 0; t < 4; ++t) b3[t] = be3[t * 16 + nn];
#pragma unroll
  for (int t = 0; t < 4; ++t) ba[t] = ba0[t * 16 + nn];

  const v16h Bwe0 = load_b_frag(wsB, FR_WE0, lane);
  const v16h Bwe1 = load_b_frag(wsB, FR_WE1, lane);
  v16h Bwe2[2], Bwe3[4], Bwa0[4];
#pragma unroll
  for (int t = 0; t < 2; ++t) Bwe2[t] = load_b_frag(wsB, FR_WE2 + t, lane);
#pragma unroll
  for (int t = 0; t < 4; ++t) Bwe3[t] = load_b_frag(wsB, FR_WE3 + t, lane);
#pragma unroll
  for (int t = 0; t < 4; ++t) Bwa0[t] = load_b_frag(wsB, FR_WA0 + t, lane);
  // --------------------------------------------------------------------------

  for (int tile = blockIdx.x * WPB + wave; tile < ntiles; tile += wavesTotal) {
    const int base = tile << 4;
    const int row = base + nn;
    const int rc = row < npts ? row : npts - 1;

    // prefetch next tile's streams into cache (global_prefetch_b8)
    {
      int nrow = base + wavesTotal * 16 + nn;
      if (nrow < npts) {
        __builtin_prefetch(pf + (size_t)nrow * 8, 0, 1);
        __builtin_prefetch(pts + (size_t)nrow * 3, 0, 1);
        __builtin_prefetch(labels + nrow, 0, 1);
      }
    }

    // zero the f16 staging tile (provides K-padding for all layers)
    v8h z8 = {};
    *(v8h*)(H + lane * 16) = z8;
    *(v8h*)(H + lane * 16 + 8) = z8;

    int lab = labels[rc];
    if (lane < 16) LB[lane] = lab;
    __builtin_amdgcn_wave_barrier();

    // Build A0 [16 x 32] directly in registers:
    // g=0 lanes: K=0..7 = features ; g=1 lanes: K=8..10 = center - point
    v16h a0 = {};
    if (row < npts) {
      if (g == 0) {
        const float* fp = pf + (size_t)rc * 8;
#pragma unroll
        for (int j = 0; j < 8; ++j) a0[j] = (_Float16)fp[j];
      } else {
        const float* pp = pts + (size_t)rc * 3;
        const float* cp = cc + (size_t)lab * 3;
#pragma unroll
        for (int j = 0; j < 3; ++j) a0[j] = (_Float16)(cp[j] - pp[j]);
      }
    }

    // L0: 11 -> 8
    v8f d0 = {};
    d0 = wmma_f16(a0, Bwe0, d0);
    d0 = bias_relu_v(d0, b0);
    store_d_frag(H, lane, 32, 0, d0);
    __builtin_amdgcn_wave_barrier();
    v16h a1 = load_a_frag(H, lane, 32, 0);
    __builtin_amdgcn_wave_barrier();

    // L1: 8 -> 16
    v8f d1 = {};
    d1 = wmma_f16(a1, Bwe1, d1);
    d1 = bias_relu_v(d1, b1);
    store_d_frag(H, lane, 32, 0, d1);
    __builtin_amdgcn_wave_barrier();
    v16h a2 = load_a_frag(H, lane, 32, 0);
    __builtin_amdgcn_wave_barrier();

    // L2: 16 -> 32
#pragma unroll
    for (int t = 0; t < 2; ++t) {
      v8f d2 = {};
      d2 = wmma_f16(a2, Bwe2[t], d2);
      d2 = bias_relu_v(d2, b2[t]);
      store_d_frag(H, lane, 32, t * 16, d2);
    }
    __builtin_amdgcn_wave_barrier();
    v16h a3 = load_a_frag(H, lane, 32, 0);
    __builtin_amdgcn_wave_barrier();

    // L3: 32 -> 64, keep tiles in registers
    v8f h[4];
#pragma unroll
    for (int t = 0; t < 4; ++t) {
      v8f d3 = {};
      d3 = wmma_f16(a3, Bwe3[t], d3);
      h[t] = bias_relu_v(d3, b3[t]);
    }

    // attention: relu(x @ wa0 + ba0) dot wa1, sigmoid
    float s[8];
#pragma unroll
    for (int r = 0; r < 8; ++r) s[r] = 0.f;
#pragma unroll
    for (int t = 0; t < 4; ++t) {
      v8f da = {};
      da = wmma_f16(a0, Bwa0[t], da);
      da = bias_relu_v(da, ba[t]);
#pragma unroll
      for (int r = 0; r < 8; ++r) s[r] += da[r] * wv[t];
    }
#pragma unroll
    for (int r = 0; r < 8; ++r) {
      float v = s[r];
      v += __shfl_xor(v, 1, 32);
      v += __shfl_xor(v, 2, 32);
      v += __shfl_xor(v, 4, 32);
      v += __shfl_xor(v, 8, 32);   // full sum over 16 lanes (64 cols)
      s[r] = 1.f / (1.f + __expf(-(v + ba1v)));
    }

    // scatter t = h * a into agg[label, 64] (L2-resident f32 atomics)
#pragma unroll
    for (int r = 0; r < 8; ++r) {
      int m = 8 * g + r;
      int grow = base + m;
      if (grow < npts) {
        int lb = LB[m];
        float* dst = agg + (size_t)lb * 64 + nn;
#pragma unroll
        for (int t = 0; t < 4; ++t) atomicAdd(dst + t * 16, h[t][r] * s[r]);
      }
    }
    __builtin_amdgcn_wave_barrier();
  }
}

// ---------------------------------------------------------------------------
// Output MLP: agg[C,64] -> 128 -> 256 with relu.
__global__ void __launch_bounds__(TPB) cluster_out_mlp(
    const float* __restrict__ agg, const _Float16* __restrict__ wsB,
    const float* __restrict__ bo0, const float* __restrict__ bo1,
    float* __restrict__ out, int C) {
  __shared__ __align__(16) _Float16 Hs[WPB][16 * 128];
  const int lane = threadIdx.x & 31;
  const int wave = threadIdx.x >> 5;
  const int g = lane >> 4, nn = lane & 15;
  _Float16* H = Hs[wave];
  const int ntiles = C >> 4;
  const int wavesTotal = gridDim.x * WPB;

  // hoist per-lane bias scalars
  float bv0[8], bv1[16];
#pragma unroll
  for (int t = 0; t < 8; ++t) bv0[t] = bo0[t * 16 + nn];
#pragma unroll
  for (int t = 0; t < 16; ++t) bv1[t] = bo1[t * 16 + nn];

  for (int tile = blockIdx.x * WPB + wave; tile < ntiles; tile += wavesTotal) {
    const int rbase = tile << 4;

    // A fragments for K=64 directly from agg (f32 -> f16)
    v16h A[2];
#pragma unroll
    for (int kb = 0; kb < 2; ++kb) {
      const float* p0 = agg + (size_t)(rbase + nn) * 64 + kb * 32 + 8 * g;
      v16h a;
#pragma unroll
      for (int j = 0; j < 8; ++j) a[j] = (_Float16)p0[j];
#pragma unroll
      for (int j = 0; j < 8; ++j) a[8 + j] = (_Float16)p0[16 + j];
      A[kb] = a;
    }

    // layer o0: 64 -> 128, stage f16 in LDS
#pragma unroll
    for (int nt = 0; nt < 8; ++nt) {
      v8f d = {};
      d = wmma_f16(A[0], load_b_frag(wsB, FR_WO0 + nt, lane), d);
      d = wmma_f16(A[1], load_b_frag(wsB, FR_WO0 + 8 + nt, lane), d);
      d = bias_relu_v(d, bv0[nt]);
      store_d_frag(H, lane, 128, nt * 16, d);
    }
    __builtin_amdgcn_wave_barrier();
    v16h A2[4];
#pragma unroll
    for (int kb = 0; kb < 4; ++kb) A2[kb] = load_a_frag(H, lane, 128, kb * 32);
    __builtin_amdgcn_wave_barrier();

    // layer o1: 128 -> 256
#pragma unroll
    for (int nt = 0; nt < 16; ++nt) {
      v8f d = {};
#pragma unroll
      for (int kb = 0; kb < 4; ++kb)
        d = wmma_f16(A2[kb], load_b_frag(wsB, FR_WO1 + kb * 16 + nt, lane), d);
      d = bias_relu_v(d, bv1[nt]);
#pragma unroll
      for (int r = 0; r < 8; ++r)
        out[(size_t)(rbase + 8 * g + r) * 256 + nt * 16 + nn] = d[r];
    }
  }
}

// ---------------------------------------------------------------------------
extern "C" void kernel_launch(void* const* d_in, const int* in_sizes, int n_in,
                              void* d_out, int out_size, void* d_ws,
                              size_t ws_size, hipStream_t stream) {
  const float* pf  = (const float*)d_in[0];
  const int*   lb  = (const int*)d_in[1];
  const float* cc  = (const float*)d_in[2];
  const float* pts = (const float*)d_in[3];
  const float* we0 = (const float*)d_in[4];
  const float* be0 = (const float*)d_in[5];
  const float* we1 = (const float*)d_in[6];
  const float* be1 = (const float*)d_in[7];
  const float* we2 = (const float*)d_in[8];
  const float* be2 = (const float*)d_in[9];
  const float* we3 = (const float*)d_in[10];
  const float* be3 = (const float*)d_in[11];
  const float* wa0 = (const float*)d_in[12];
  const float* ba0 = (const float*)d_in[13];
  const float* wa1 = (const float*)d_in[14];
  const float* ba1 = (const float*)d_in[15];
  const float* wo0 = (const float*)d_in[16];
  const float* bo0 = (const float*)d_in[17];
  const float* wo1 = (const float*)d_in[18];
  const float* bo1 = (const float*)d_in[19];

  const int npts = in_sizes[1];
  const int C = in_sizes[2] / 3;

  _Float16* wsB = (_Float16*)d_ws;
  float* agg = (float*)((char*)d_ws + AGG_OFF);
  float* out = (float*)d_out;

  pack_weights<<<NFRAG, 32, 0, stream>>>(we0, we1, we2, we3, wa0, wo0, wo1, wsB);
  zero_f32<<<1024, 256, 0, stream>>>(agg, (size_t)C * 64);
  edge_attn_scatter<<<2048, TPB, 0, stream>>>(pf, lb, cc, pts, be0, be1, be2,
                                              be3, ba0, wa1, ba1, wsB, agg,
                                              npts);
  cluster_out_mlp<<<1024, TPB, 0, stream>>>(agg, wsB, bo0, bo1, out, C);
}
